// GNNFilm_43258910605914
// MI455X (gfx1250) — matrix operand order repaired and checked by reading, not compiled, not run on hardware
//
#include <hip/hip_runtime.h>
#include <hip/hip_bf16.h>

// ---------------------------------------------------------------------------
// GNN-FiLM on MI455X (gfx1250): bf16 WMMA GEMMs + L2-resident edge scatter.
// ---------------------------------------------------------------------------

typedef __bf16 bf16_t;
typedef __attribute__((ext_vector_type(16))) __bf16 v16bf;
typedef __attribute__((ext_vector_type(8)))  float  v8f;
typedef int v4i_g __attribute__((vector_size(16)));             // builtin pointee
typedef __attribute__((address_space(3))) v4i_g* lds_v4i_p;     // LDS-side pointer

#define NN 50000
#define EE 800000
#define DH 128
#define RR 4
#define TCOLS 1920            // 128 skip | 256 film_skip | 512 rel | 1024 film
#define COLTILES 120          // TCOLS/16
#define PACK_PER_LAYER (COLTILES * 4 * 32 * 16)   // 245760 bf16 per layer

#if __has_builtin(__builtin_amdgcn_global_load_async_to_lds_b128)
#define USE_ASYNC_LDS 1
#else
#define USE_ASYNC_LDS 0
#endif

// ---------------------------------------------------------------------------
// Weight packing into WMMA B-operand layout.
// B (32x16, KxN) in this code's contiguous k order: lanes 0-15 -> K=0..15, lanes 16-31 -> K=16..31,
// element e in lane's v16bf -> k = kc*32 + (lane>>4)*16 + e, n = tile*16+(lane&15).
// ---------------------------------------------------------------------------
__global__ void __launch_bounds__(256)
pack_weights_kernel(const float* __restrict__ Wskip, const float* __restrict__ Wfsk,
                    const float* __restrict__ Wrel,  const float* __restrict__ Wfilm,
                    bf16_t* __restrict__ Wpack)
{
    int idx = blockIdx.x * blockDim.x + threadIdx.x;
    if (idx >= 3 * PACK_PER_LAYER) return;
    int l    = idx / PACK_PER_LAYER;
    int rem  = idx % PACK_PER_LAYER;
    int e    = rem & 15;
    int lane = (rem >> 4) & 31;
    int kc   = (rem >> 9) & 3;
    int tile = rem >> 11;                       // 0..119
    int n = tile * 16 + (lane & 15);            // 0..1919
    int k = kc * 32 + ((lane >> 4) << 4) + e;   // 0..127
    float w;
    if (n < 128) {
        w = Wskip[((size_t)l * 128 + k) * 128 + n];
    } else if (n < 384) {
        w = Wfsk[((size_t)l * 128 + k) * 256 + (n - 128)];
    } else if (n < 896) {
        int r = (n - 384) >> 7, cc = (n - 384) & 127;
        w = Wrel[(((size_t)l * RR + r) * 128 + k) * 128 + cc];
    } else {
        int r = (n - 896) >> 8, cc = (n - 896) & 255;
        w = Wfilm[(((size_t)l * RR + r) * 128 + k) * 256 + cc];
    }
    Wpack[idx] = (bf16_t)w;
}

__global__ void __launch_bounds__(256)
pack_lin1_kernel(const float* __restrict__ L1w, bf16_t* __restrict__ L1pack)
{
    int idx = blockIdx.x * blockDim.x + threadIdx.x;
    if (idx >= 2 * 4 * 32 * 16) return;
    int e    = idx & 15;
    int lane = (idx >> 4) & 31;
    int kc   = (idx >> 9) & 3;
    int tile = idx >> 11;                      // 0..1
    int n = tile * 16 + (lane & 15);           // 0..31
    int k = kc * 32 + ((lane >> 4) << 4) + e;  // 0..127
    L1pack[idx] = (bf16_t)L1w[(size_t)k * 32 + n];
}

__global__ void __launch_bounds__(256)
cvt_bf16_kernel(const float* __restrict__ x, bf16_t* __restrict__ xbf, int total)
{
    int idx = blockIdx.x * blockDim.x + threadIdx.x;
    if (idx < total) xbf[idx] = (bf16_t)x[idx];
}

// ---------------------------------------------------------------------------
// GEMM: C[M x (numColTiles*16)] = Abf16[M x 128] @ Bpack, via v_wmma bf16.
// Block = 256 threads = 8 waves. A tile (16 rows x 128 K, bf16) staged in LDS
// (async global->LDS on gfx1250 when available). Each wave owns one 16x16
// output tile; K loop = 4 chunks of 32. A fragment uses the same contiguous k order as Bpack.
// LDC / bias / activation are compile-time to keep the epilogue branch-free.
// ---------------------------------------------------------------------------
template <int LDC, bool HAS_BIAS, int ACT>
__global__ void __launch_bounds__(256)
gemm_bf16_kernel(const bf16_t* __restrict__ A, const bf16_t* __restrict__ Bpack,
                 float* __restrict__ C, int numColTiles,
                 const float* __restrict__ bias)
{
    __shared__ uint4 lds4[256];                  // 16 rows x 128 bf16 = 4 KB
    unsigned* lds = reinterpret_cast<unsigned*>(lds4);
    const int t  = threadIdx.x;
    const int mt = blockIdx.x;

    // Cooperative A-tile load: 256 threads x 16B = 4 KB.
    {
        int row = t >> 4;                        // 0..15
        int c16 = t & 15;                        // 16B chunk within 256B row
        const uint4* gp = reinterpret_cast<const uint4*>(A + (size_t)(mt * 16 + row) * DH) + c16;
#if USE_ASYNC_LDS
        // gfx1250 GLOBAL_LOAD_ASYNC_TO_LDS_B128: memory -> LDS, ASYNCcnt-tracked.
        __builtin_amdgcn_global_load_async_to_lds_b128(
            (v4i_g*)gp, (lds_v4i_p)&lds4[t], 0, 0);
        asm volatile("s_wait_asynccnt 0" ::: "memory");
#else
        lds4[t] = *gp;
#endif
    }
    __syncthreads();

    const int wid  = t >> 5;
    const int lane = t & 31;
    const int nt   = blockIdx.y * 8 + wid;
    if (nt >= numColTiles) return;

    const int m = lane & 15;
    const int h = lane >> 4;

    v8f c = {0.f, 0.f, 0.f, 0.f, 0.f, 0.f, 0.f, 0.f};
#pragma unroll
    for (int kc = 0; kc < 4; ++kc) {
        // A fragment: pair p -> k = kc*32 + h*16 + p*2 (+0/+1)
        union { v16bf v; unsigned u[8]; } a;
#pragma unroll
        for (int p = 0; p < 8; ++p) {
            int k = kc * 32 + (h << 4) + (p << 1);
            a.u[p] = lds[m * 64 + (k >> 1)];
        }
        // B fragment: contiguous 32B per lane from packed weights.
        const v16bf* bp = reinterpret_cast<const v16bf*>(
            Bpack + ((size_t)(nt * 4 + kc) * 32 + lane) * 16);
        v16bf b = *bp;
        c = __builtin_amdgcn_wmma_f32_16x16x32_bf16(
                false, a.v, false, b, (short)0, c, false, false);
    }

    // C/D layout: VGPR i -> row (i + h*8), col = lane&15.
    const int colBase = nt * 16 + m;
    const int rowBase = mt * 16 + h * 8;
    float bval = 0.f;
    if (HAS_BIAS) bval = bias[colBase];
    float* cp = C + (size_t)rowBase * LDC + colBase;
#pragma unroll
    for (int i = 0; i < 8; ++i) {
        float v = c[i];
        if (HAS_BIAS) v += bval;
        if (ACT == 1) v = v > 0.f ? v : 0.2f * v;
        cp[i * LDC] = v;
    }
}

// ---------------------------------------------------------------------------
// Edge phase: wave per edge. msg = relu(gamma[dst,r] * xr[src,r] + beta[dst,r]);
// atomic segment-sum into (dst,r) accumulators (102 MB -> resident in 192MB L2).
// ---------------------------------------------------------------------------
__global__ void __launch_bounds__(256)
edge_kernel(const int* __restrict__ ei, const int* __restrict__ et,
            const float* __restrict__ T, float* __restrict__ sums,
            float* __restrict__ cnts)
{
    int wid  = (blockIdx.x * blockDim.x + threadIdx.x) >> 5;
    int lane = threadIdx.x & 31;
    if (wid >= EE) return;
    int s = ei[wid];
    int d = ei[EE + wid];
    int r = et[wid];
    const float4* xr = reinterpret_cast<const float4*>(T + (size_t)s * TCOLS + 384 + r * 128);
    const float4* bg = reinterpret_cast<const float4*>(T + (size_t)d * TCOLS + 896 + r * 256);
    float4 xv = xr[lane];
    float4 bv = bg[lane];        // beta  = first 128 of film output
    float4 gv = bg[lane + 32];   // gamma = second 128
    float4 mv;
    mv.x = fmaxf(0.f, gv.x * xv.x + bv.x);
    mv.y = fmaxf(0.f, gv.y * xv.y + bv.y);
    mv.z = fmaxf(0.f, gv.z * xv.z + bv.z);
    mv.w = fmaxf(0.f, gv.w * xv.w + bv.w);
    float* sp = sums + ((size_t)d * RR + r) * DH + lane * 4;
    atomicAdd(sp + 0, mv.x);
    atomicAdd(sp + 1, mv.y);
    atomicAdd(sp + 2, mv.z);
    atomicAdd(sp + 3, mv.w);
    if (lane == 0) atomicAdd(&cnts[d * RR + r], 1.0f);
}

// Self path + per-relation means -> pre-BN activations. Wave per node.
__global__ void __launch_bounds__(256)
aggregate_kernel(const float* __restrict__ T, const float* __restrict__ sums,
                 const float* __restrict__ cnts, float* __restrict__ hpre)
{
    int wid  = (blockIdx.x * blockDim.x + threadIdx.x) >> 5;
    int lane = threadIdx.x & 31;
    if (wid >= NN) return;
    const float4* t4 = reinterpret_cast<const float4*>(T + (size_t)wid * TCOLS);
    float4 sk = t4[lane];        // skip transform
    float4 be = t4[32 + lane];   // beta_s
    float4 ga = t4[64 + lane];   // gamma_s
    float4 acc;
    acc.x = fmaxf(0.f, ga.x * sk.x + be.x);
    acc.y = fmaxf(0.f, ga.y * sk.y + be.y);
    acc.z = fmaxf(0.f, ga.z * sk.z + be.z);
    acc.w = fmaxf(0.f, ga.w * sk.w + be.w);
#pragma unroll
    for (int r = 0; r < RR; ++r) {
        float inv = 1.0f / fmaxf(cnts[(size_t)wid * RR + r], 1.0f);
        const float4* sv = reinterpret_cast<const float4*>(sums + ((size_t)wid * RR + r) * DH);
        float4 s = sv[lane];
        acc.x += s.x * inv; acc.y += s.y * inv;
        acc.z += s.z * inv; acc.w += s.w * inv;
    }
    reinterpret_cast<float4*>(hpre + (size_t)wid * DH)[lane] = acc;
}

// Per-channel sum / sumsq (block-partial in registers, one atomic per block).
__global__ void __launch_bounds__(128)
bn_stats_kernel(const float* __restrict__ h, float* __restrict__ stat)
{
    int c = threadIdx.x;
    float s = 0.f, q = 0.f;
    for (int n = blockIdx.x; n < NN; n += gridDim.x) {
        float v = h[(size_t)n * DH + c];
        s += v; q += v * v;
    }
    atomicAdd(&stat[c], s);
    atomicAdd(&stat[DH + c], q);
}

// Normalize + affine, emit bf16 activations for the next GEMM.
__global__ void __launch_bounds__(256)
bn_apply_kernel(const float* __restrict__ h, const float* __restrict__ stat,
                const float* __restrict__ w, const float* __restrict__ b,
                bf16_t* __restrict__ xbf, int total)
{
    int idx = blockIdx.x * blockDim.x + threadIdx.x;
    if (idx >= total) return;
    int c = idx & (DH - 1);
    const float invN = 1.0f / (float)NN;
    float mu  = stat[c] * invN;
    float var = stat[DH + c] * invN - mu * mu;
    float xn = (h[idx] - mu) * rsqrtf(var + 1e-5f) * w[c] + b[c];
    xbf[idx] = (bf16_t)xn;
}

// Tiny 32->40 tail.
__global__ void __launch_bounds__(256)
lin2_kernel(const float* __restrict__ mid, const float* __restrict__ w,
            const float* __restrict__ b, float* __restrict__ out, int total)
{
    int idx = blockIdx.x * blockDim.x + threadIdx.x;
    if (idx >= total) return;
    int n = idx / 40;
    int j = idx - n * 40;
    float acc = b[j];
    const float* m = mid + (size_t)n * 32;
#pragma unroll
    for (int c2 = 0; c2 < 32; ++c2) acc += m[c2] * w[c2 * 40 + j];
    out[idx] = acc;
}

// ---------------------------------------------------------------------------
extern "C" void kernel_launch(void* const* d_in, const int* in_sizes, int n_in,
                              void* d_out, int out_size, void* d_ws, size_t ws_size,
                              hipStream_t stream)
{
    const float* x     = (const float*)d_in[0];
    const int*   ei    = (const int*)d_in[1];
    const int*   et    = (const int*)d_in[2];
    const float* Wskip = (const float*)d_in[3];
    const float* Wfsk  = (const float*)d_in[4];
    const float* Wrel  = (const float*)d_in[5];
    const float* Wfilm = (const float*)d_in[6];
    const float* bnw   = (const float*)d_in[7];
    const float* bnb   = (const float*)d_in[8];
    const float* l1w   = (const float*)d_in[9];
    const float* l1b   = (const float*)d_in[10];
    const float* l2w   = (const float*)d_in[11];
    const float* l2b   = (const float*)d_in[12];
    float* out = (float*)d_out;

    char* ws = (char*)d_ws;
    size_t off = 0;
    auto alloc = [&](size_t bytes) -> void* {
        void* p = ws + off;
        off += bytes;
        off = (off + 255) & ~(size_t)255;
        return p;
    };
    bf16_t* Xbf    = (bf16_t*)alloc((size_t)NN * DH * 2);
    float*  T      = (float*) alloc((size_t)NN * TCOLS * 4);
    float*  sums   = (float*) alloc((size_t)NN * RR * DH * 4);
    float*  cnts   = (float*) alloc((size_t)NN * RR * 4);
    float*  hpre   = (float*) alloc((size_t)NN * DH * 4);
    float*  stat   = (float*) alloc(2 * DH * 4);
    bf16_t* Wpack  = (bf16_t*)alloc((size_t)3 * PACK_PER_LAYER * 2);
    bf16_t* L1pack = (bf16_t*)alloc((size_t)4096 * 2);
    float*  mid    = (float*) alloc((size_t)NN * 32 * 4);

    // One-time-per-call weight packing (deterministic).
    pack_weights_kernel<<<(3 * PACK_PER_LAYER + 255) / 256, 256, 0, stream>>>(
        Wskip, Wfsk, Wrel, Wfilm, Wpack);
    pack_lin1_kernel<<<16, 256, 0, stream>>>(l1w, L1pack);
    cvt_bf16_kernel<<<(NN * DH + 255) / 256, 256, 0, stream>>>(x, Xbf, NN * DH);

    const int MTILES = NN / 16;   // 3125, exact
    for (int l = 0; l < 3; ++l) {
        (void)hipMemsetAsync(sums, 0, (size_t)NN * RR * DH * 4, stream);
        (void)hipMemsetAsync(cnts, 0, (size_t)NN * RR * 4, stream);
        (void)hipMemsetAsync(stat, 0, 2 * DH * 4, stream);

        // Fused layer GEMM: [N,128] x [128,1920] -> T
        gemm_bf16_kernel<TCOLS, false, 0>
            <<<dim3(MTILES, COLTILES / 8), 256, 0, stream>>>(
                Xbf, Wpack + (size_t)l * PACK_PER_LAYER, T, COLTILES, nullptr);

        edge_kernel<<<(EE * 32 + 255) / 256, 256, 0, stream>>>(ei, et, T, sums, cnts);
        aggregate_kernel<<<(NN * 32 + 255) / 256, 256, 0, stream>>>(T, sums, cnts, hpre);
        bn_stats_kernel<<<512, 128, 0, stream>>>(hpre, stat);
        bn_apply_kernel<<<(NN * DH + 255) / 256, 256, 0, stream>>>(
            hpre, stat, bnw + l * DH, bnb + l * DH, Xbf, NN * DH);
    }

    // Head: mid = leaky_relu(h @ lin1 + b1) via WMMA, then tiny lin2.
    gemm_bf16_kernel<32, true, 1><<<dim3(MTILES, 1), 256, 0, stream>>>(
        Xbf, L1pack, mid, 2, l1b);
    lin2_kernel<<<(NN * 40 + 255) / 256, 256, 0, stream>>>(mid, l2w, l2b, out, NN * 40);
}
